// HeteroGNN_41540923686987
// MI455X (gfx1250) — matrix operand order repaired and checked by reading, not compiled, not run on hardware
//
#include <hip/hip_runtime.h>

// ---------- types ----------
typedef __attribute__((ext_vector_type(16))) __bf16 v16bf;
typedef __attribute__((ext_vector_type(8)))  __bf16 v8bf;
typedef __attribute__((ext_vector_type(8)))  float  v8f;

// f32 -> bf16 round-to-nearest-even (bit trick; avoids libcalls)
__device__ __forceinline__ __bf16 f2bf(float f) {
    union { float f; unsigned u; } x; x.f = f;
    unsigned r = x.u + 0x7FFFu + ((x.u >> 16) & 1u);
    unsigned short h = (unsigned short)(r >> 16);
    __bf16 b;
    __builtin_memcpy(&b, &h, 2);
    return b;
}

// ---------- edge scatter: one wave32 per edge, float4 per lane ----------
// agg[dst] += x[src]; cnt[dst] += 1
__global__ __launch_bounds__(256) void scatter_mean_accum(
        const float* __restrict__ x, const int* __restrict__ ei,
        float* __restrict__ agg, float* __restrict__ cnt, int E) {
    int tid  = blockIdx.x * blockDim.x + threadIdx.x;
    int lane = tid & 31;
    int e    = tid >> 5;
    if (e >= E) return;
    int src = ei[e];
    int dst = ei[E + e];
    const float4 v = *(const float4*)(x + (size_t)src * 128 + lane * 4);
    float* dp = agg + (size_t)dst * 128 + lane * 4;
    atomicAdd(dp + 0, v.x);
    atomicAdd(dp + 1, v.y);
    atomicAdd(dp + 2, v.z);
    atomicAdd(dp + 3, v.w);
    if (lane == 0) atomicAdd(cnt + dst, 1.0f);
}

// ---------- WMMA GEMM: Out[M x 128] (+)= rowscale(A[M x 128]) @ W[128 x 128] ----------
// block = 256 threads (8 waves), tile = 128 rows x 128 cols, K-loop 4 x 16x16x32 bf16
__global__ __launch_bounds__(256) void gemm_tile_wmma(
        const float* __restrict__ A, const float* __restrict__ rowcnt,
        const float* __restrict__ W, float* __restrict__ Out, int M, int beta) {
    // B pre-swizzled so each b-fragment is one contiguous 32B LDS read:
    // Wl[ks][nt][lane][e] = bf16( W[ ks*32 + (lane>>4)*16 + e ][ nt*16 + (lane&15) ] )
    __shared__ __align__(32) __bf16 Wl[4][8][32][16];   // 32 KB
    __shared__ __align__(16) __bf16 Al[8][16][128];     // 32 KB, row-major per wave
    const int tid  = threadIdx.x;
    const int wave = tid >> 5;
    const int lane = tid & 31;
    const int rowBase = blockIdx.x * 128;

    // stage + swizzle W (f32 -> bf16)
    for (int idx = tid; idx < 16384; idx += 256) {
        int e  = idx & 15;
        int l  = (idx >> 4) & 31;
        int nt = (idx >> 9) & 7;
        int ks = idx >> 12;
        int k  = ks * 32 + (l >> 4) * 16 + e;
        int n  = nt * 16 + (l & 15);
        Wl[ks][nt][l][e] = f2bf(W[k * 128 + n]);
    }

    // stage this wave's 16 A rows (fused 1/cnt scaling for segment-mean)
    for (int i = lane; i < 512; i += 32) {            // 512 float4 = 16x128 floats
        int r = i >> 5;
        int c = (i & 31) << 2;
        int row = rowBase + wave * 16 + r;
        float4 v = make_float4(0.f, 0.f, 0.f, 0.f);
        float  s = 1.0f;
        if (row < M) {
            v = *(const float4*)(A + (size_t)row * 128 + c);
            if (rowcnt) s = 1.0f / fmaxf(rowcnt[row], 1.0f);
        }
        Al[wave][r][c + 0] = f2bf(v.x * s);
        Al[wave][r][c + 1] = f2bf(v.y * s);
        Al[wave][r][c + 2] = f2bf(v.z * s);
        Al[wave][r][c + 3] = f2bf(v.w * s);
    }
    __syncthreads();

    v8f acc[8] = {};
    const int arow = lane & 15;
    const int koff = (lane >> 4) << 4;   // A: lanes>=16 hold K 16..31
    #pragma unroll
    for (int ks = 0; ks < 4; ++ks) {
        // A fragment in contiguous K order (k = 16*(lane>>4) + e), matching Wl: two adjacent 16B chunks
        v8bf alo = *(const v8bf*)&Al[wave][arow][ks * 32 + koff];
        v8bf ahi = *(const v8bf*)&Al[wave][arow][ks * 32 + 8 + koff];
        v16bf a;
        #pragma unroll
        for (int i = 0; i < 8; ++i) { a[i] = alo[i]; a[i + 8] = ahi[i]; }
        #pragma unroll
        for (int nt = 0; nt < 8; ++nt) {
            v16bf b = *(const v16bf*)&Wl[ks][nt][lane][0];
            acc[nt] = __builtin_amdgcn_wmma_f32_16x16x32_bf16(
                false, a, false, b, (short)0, acc[nt], false, false);
        }
    }

    // C/D layout: VGPR r -> M=r (lanes 0-15) / M=r+8 (lanes 16-31), N = lane&15
    const int m0 = (lane >> 4) << 3;
    const int n  = lane & 15;
    #pragma unroll
    for (int nt = 0; nt < 8; ++nt) {
        #pragma unroll
        for (int r = 0; r < 8; ++r) {
            int row = rowBase + wave * 16 + m0 + r;
            if (row < M) {
                float* p = Out + (size_t)row * 128 + nt * 16 + n;
                float v = acc[nt][r];
                if (beta) v += *p;
                *p = v;
            }
        }
    }
}

// ---------- bias (+optional second bias) + ReLU ----------
__global__ __launch_bounds__(256) void bias_relu(
        float* __restrict__ x, const float* __restrict__ b1,
        const float* __restrict__ b2, int M) {
    long long i = (long long)blockIdx.x * blockDim.x + threadIdx.x;
    if (i >= (long long)M * 128) return;
    int c = (int)(i & 127);
    float v = x[i] + b1[c];
    if (b2) v += b2[c];
    x[i] = fmaxf(v, 0.0f);
}

// ---------- global mean pool accumulate: wave32 per node ----------
__global__ __launch_bounds__(256) void pool_accum(
        const float* __restrict__ xa, const int* __restrict__ batch,
        float* __restrict__ pooled, float* __restrict__ pcnt, int n) {
    int tid  = blockIdx.x * blockDim.x + threadIdx.x;
    int lane = tid & 31;
    int node = tid >> 5;
    if (node >= n) return;
    int g = batch[node];
    const float4 v = *(const float4*)(xa + (size_t)node * 128 + lane * 4);
    float* dp = pooled + g * 128 + lane * 4;
    atomicAdd(dp + 0, v.x);
    atomicAdd(dp + 1, v.y);
    atomicAdd(dp + 2, v.z);
    atomicAdd(dp + 3, v.w);
    if (lane == 0) atomicAdd(pcnt + g, 1.0f);
}

// ---------- final head: [64 x 128] @ [128 x 10] + b ----------
__global__ __launch_bounds__(256) void final_out(
        const float* __restrict__ pooled, const float* __restrict__ pcnt,
        const float* __restrict__ Wo, const float* __restrict__ bo,
        float* __restrict__ out) {
    int t = blockIdx.x * blockDim.x + threadIdx.x;
    if (t >= 64 * 10) return;
    int g = t / 10, c = t % 10;
    float s = 0.0f;
    for (int k = 0; k < 128; ++k) s += pooled[g * 128 + k] * Wo[k * 10 + c];
    out[t] = s / fmaxf(pcnt[g], 1.0f) + bo[c];
}

extern "C" void kernel_launch(void* const* d_in, const int* in_sizes, int n_in,
                              void* d_out, int out_size, void* d_ws, size_t ws_size,
                              hipStream_t stream) {
    (void)n_in; (void)out_size; (void)ws_size;
    const float* x_a   = (const float*)d_in[0];
    const float* x_b   = (const float*)d_in[1];
    const int*   ei_aa = (const int*)d_in[2];
    const int*   ei_ab = (const int*)d_in[3];
    const int*   ei_ba = (const int*)d_in[4];
    const int*   batch = (const int*)d_in[5];
    const float* W_out = (const float*)d_in[24];
    const float* b_out = (const float*)d_in[25];

    const int nA  = in_sizes[0] / 128;
    const int nB  = in_sizes[1] / 128;
    const int Eaa = in_sizes[2] / 2;
    const int Eab = in_sizes[3] / 2;
    const int Eba = in_sizes[4] / 2;
    const size_t maxN = (size_t)(nA > nB ? nA : nB);

    // workspace layout (f32): cnt directly follows agg, pcnt follows pooled
    float* p = (float*)d_ws;
    float* bufA0  = p; p += (size_t)nA * 128;
    float* bufB0  = p; p += (size_t)nB * 128;
    float* bufA1  = p; p += (size_t)nA * 128;
    float* bufB1  = p; p += (size_t)nB * 128;
    float* agg    = p; p += maxN * 128;
    float* cnt    = p; p += maxN;
    float* pooled = p; p += 64 * 128;
    float* pcnt   = p; p += 64;

    auto seg_mean_into = [&](const float* xsrc, const int* ei, int E) {
        hipMemsetAsync(agg, 0, (maxN * 128 + maxN) * sizeof(float), stream);
        int blocks = (int)(((size_t)E * 32 + 255) / 256);
        scatter_mean_accum<<<blocks, 256, 0, stream>>>(xsrc, ei, agg, cnt, E);
    };
    auto gemm = [&](const float* A, const float* rc, const float* W,
                    float* O, int M, int beta) {
        gemm_tile_wmma<<<(M + 127) / 128, 256, 0, stream>>>(A, rc, W, O, M, beta);
    };

    for (int layer = 0; layer < 2; ++layer) {
        int wb = 6 + layer * 9;
        const float* Wn_aa = (const float*)d_in[wb + 0];
        const float* bn_aa = (const float*)d_in[wb + 1];
        const float* Wr_aa = (const float*)d_in[wb + 2];
        const float* Wn_ab = (const float*)d_in[wb + 3];
        const float* bn_ab = (const float*)d_in[wb + 4];
        const float* Wr_ab = (const float*)d_in[wb + 5];
        const float* Wn_ba = (const float*)d_in[wb + 6];
        const float* bn_ba = (const float*)d_in[wb + 7];
        const float* Wr_ba = (const float*)d_in[wb + 8];

        const float* xa = (layer == 0) ? x_a : bufA0;
        const float* xb = (layer == 0) ? x_b : bufB0;
        float* oa = (layer == 0) ? bufA0 : bufA1;
        float* ob = (layer == 0) ? bufB0 : bufB1;

        // out_a = relu( mean_aa@Wn_aa + bn_aa + xa@Wr_aa + mean_ba@Wn_ba + bn_ba + xa@Wr_ba )
        seg_mean_into(xa, ei_aa, Eaa);
        gemm(agg, cnt,     Wn_aa, oa, nA, 0);
        gemm(xa,  nullptr, Wr_aa, oa, nA, 1);
        seg_mean_into(xb, ei_ba, Eba);
        gemm(agg, cnt,     Wn_ba, oa, nA, 1);
        gemm(xa,  nullptr, Wr_ba, oa, nA, 1);
        bias_relu<<<(int)(((size_t)nA * 128 + 255) / 256), 256, 0, stream>>>(oa, bn_aa, bn_ba, nA);

        // out_b = relu( mean_ab@Wn_ab + bn_ab + xb@Wr_ab )
        seg_mean_into(xa, ei_ab, Eab);
        gemm(agg, cnt,     Wn_ab, ob, nB, 0);
        gemm(xb,  nullptr, Wr_ab, ob, nB, 1);
        bias_relu<<<(int)(((size_t)nB * 128 + 255) / 256), 256, 0, stream>>>(ob, bn_ab, nullptr, nB);
    }

    // global mean pool over node type 'a' + classifier head
    hipMemsetAsync(pooled, 0, (64 * 128 + 64) * sizeof(float), stream);
    pool_accum<<<(int)(((size_t)nA * 32 + 255) / 256), 256, 0, stream>>>(bufA1, batch, pooled, pcnt, nA);
    final_out<<<3, 256, 0, stream>>>(pooled, pcnt, W_out, b_out, (float*)d_out);
}
